// ExpertParallelFrontBlockDS_2834678415771
// MI455X (gfx1250) — compile-verified, matrix operand order checked by reading, not run on hardware
//
#include <hip/hip_runtime.h>
#include <hip/hip_bf16.h>

#define D_MODEL 2048
#define D_FF    8192
#define NEXP    8
#define CAP     1024
#define NTOK    8192

typedef __attribute__((ext_vector_type(16))) __bf16 v16bf;
typedef __attribute__((ext_vector_type(8)))  float  v8f;

#define LDS_STRIDE 40  // halves per LDS tile row: 32 data + 8 pad (80B, 16B-aligned, bank-conflict free)
#define TILE_HALVES (128 * LDS_STRIDE)

__device__ __forceinline__ unsigned int pack_bf16x2(float lo, float hi) {
    unsigned short l = __builtin_bit_cast(unsigned short, (__bf16)lo);
    unsigned short h = __builtin_bit_cast(unsigned short, (__bf16)hi);
    return (unsigned int)l | ((unsigned int)h << 16);
}

// CDNA5 async global->LDS copy (32B per call as 2x B128), tracked with ASYNCcnt.
__device__ __forceinline__ void async_copy32_to_lds(const unsigned short* g, unsigned short* ldsp) {
    unsigned l0 = (unsigned)(uintptr_t)ldsp;   // flat LDS addr[31:0] == LDS byte offset
    asm volatile("global_load_async_to_lds_b128 %0, %1, off"
                 :: "v"(l0), "v"(g) : "memory");
    asm volatile("global_load_async_to_lds_b128 %0, %1, off"
                 :: "v"(l0 + 16u), "v"(g + 8) : "memory");
}

__device__ __forceinline__ void wait_asynccnt0() {
    asm volatile("s_wait_asynccnt 0x0" ::: "memory");
}

// ---------------- Kernel 1: top-1 gate (argmax of logits; softmax is monotonic) ----------------
__global__ __launch_bounds__(256) void moe_gate(const float* __restrict__ x,
                                                const float* __restrict__ wg,
                                                int* __restrict__ idx) {
    const int wave = threadIdx.x >> 5;
    const int lane = threadIdx.x & 31;
    const int s = blockIdx.x * 8 + wave;      // one wave per token
    const float* xp = x + (size_t)s * D_MODEL;

    float acc[8];
#pragma unroll
    for (int e = 0; e < 8; ++e) acc[e] = 0.0f;

    for (int d = lane; d < D_MODEL; d += 32) {
        float xv = xp[d];
        const float4* wrow = (const float4*)(wg + d * 8);
        float4 w0 = wrow[0];
        float4 w1 = wrow[1];
        acc[0] += xv * w0.x; acc[1] += xv * w0.y; acc[2] += xv * w0.z; acc[3] += xv * w0.w;
        acc[4] += xv * w1.x; acc[5] += xv * w1.y; acc[6] += xv * w1.z; acc[7] += xv * w1.w;
    }
#pragma unroll
    for (int off = 16; off > 0; off >>= 1) {
#pragma unroll
        for (int e = 0; e < 8; ++e) acc[e] += __shfl_xor(acc[e], off);
    }
    if (lane == 0) {
        int best = 0;
        float bv = acc[0];
#pragma unroll
        for (int e = 1; e < 8; ++e) {
            if (acc[e] > bv) { bv = acc[e]; best = e; }  // first-max tie break (== jnp.argmax)
        }
        idx[s] = best;
    }
}

// ---------------- Kernel 2: deterministic capacity assignment via block-wide 8-way prefix scan --
__global__ __launch_bounds__(1024) void moe_assign(const int* __restrict__ idx,
                                                   int* __restrict__ slot_src) {
    __shared__ int pref[1024][8];
    const int t = threadIdx.x;

    // init gather table to "empty"
#pragma unroll
    for (int i = 0; i < 8; ++i) slot_src[t * 8 + i] = -1;

    int myidx[8];
    const int4* ip = (const int4*)(idx + t * 8);
    int4 a = ip[0], b = ip[1];
    myidx[0] = a.x; myidx[1] = a.y; myidx[2] = a.z; myidx[3] = a.w;
    myidx[4] = b.x; myidx[5] = b.y; myidx[6] = b.z; myidx[7] = b.w;

    int cnt[8];
#pragma unroll
    for (int e = 0; e < 8; ++e) cnt[e] = 0;
#pragma unroll
    for (int i = 0; i < 8; ++i) cnt[myidx[i]]++;

#pragma unroll
    for (int e = 0; e < 8; ++e) pref[t][e] = cnt[e];
    __syncthreads();

    // Hillis-Steele inclusive scan over 1024 threads, 8 experts in parallel
    for (int off = 1; off < 1024; off <<= 1) {
        int v[8];
        const bool has = (t >= off);
        if (has) {
#pragma unroll
            for (int e = 0; e < 8; ++e) v[e] = pref[t - off][e];
        }
        __syncthreads();
        if (has) {
#pragma unroll
            for (int e = 0; e < 8; ++e) pref[t][e] += v[e];
        }
        __syncthreads();
    }

    int run[8];
#pragma unroll
    for (int e = 0; e < 8; ++e) run[e] = pref[t][e] - cnt[e];  // exclusive base

#pragma unroll
    for (int i = 0; i < 8; ++i) {
        int e = myidx[i];
        int c = run[e]++;
        if (c < CAP) slot_src[e * CAP + c] = t * 8 + i;         // over-capacity tokens dropped
    }
}

// ---------------- Kernel 3: gather tokens to dense [E*C, d] bf16 dispatch buffer ---------------
__global__ __launch_bounds__(256) void moe_dispatch(const float* __restrict__ x,
                                                    const int* __restrict__ slot_src,
                                                    unsigned short* __restrict__ Abf) {
    const int r = blockIdx.x;           // slot row 0..8191
    const int t = threadIdx.x;
    const int s = slot_src[r];
    const int d0 = t * 8;

    uint4 outv = {0u, 0u, 0u, 0u};
    if (s >= 0) {
        const float4* xp = (const float4*)(x + (size_t)s * D_MODEL + d0);
        float4 f0 = xp[0];
        float4 f1 = xp[1];
        outv.x = pack_bf16x2(f0.x, f0.y);
        outv.y = pack_bf16x2(f0.z, f0.w);
        outv.z = pack_bf16x2(f1.x, f1.y);
        outv.w = pack_bf16x2(f1.z, f1.w);
    }
    *(uint4*)(Abf + (size_t)r * D_MODEL + d0) = outv;
}

// ---------------- Kernel 4: per-expert bf16 WMMA GEMM, 128x128 tile / block --------------------
// block = 256 threads = 8 waves laid out 4(M) x 2(N); each wave: 2x4 grid of 16x16 WMMA tiles.
// Double-buffered LDS; A tile staged with GLOBAL_LOAD_ASYNC_TO_LDS_B128 (ASYNCcnt), B tile
// converted fp32->bf16 through VGPRs.
__global__ __launch_bounds__(256) void moe_gemm(const unsigned short* __restrict__ Abf,
                                                const float* __restrict__ we,
                                                const float* __restrict__ be,
                                                float* __restrict__ out) {
    __shared__ unsigned short Alds[2][TILE_HALVES];
    __shared__ unsigned short Blds[2][TILE_HALVES];

    const int nt   = blockIdx.x;            // d_ff tile (64)
    const int mt   = blockIdx.y;            // capacity tile (8)
    const int e    = blockIdx.z;            // expert (8)
    const int tid  = threadIdx.x;
    const int wave = tid >> 5;
    const int lane = tid & 31;
    const int waveM = wave >> 1;            // 0..3  -> 32 rows
    const int waveN = wave & 1;             // 0..1  -> 64 cols

    const int rowBase = e * CAP + mt * 128; // row into A / out
    const int colBase = nt * 128;           // col into we / out

    v8f acc[2][4] = {};

    const bool low  = lane < 16;
    const int  rsel = low ? lane : (lane - 16);
    const int  hoff = low ? 0 : 8;          // half-offset per ISA bf16 A/B lane layout

    // A-tile fill indices: thread -> (row, 32B segment)
    const int ar    = tid >> 1;
    const int apart = tid & 1;
    // B-tile fill indices: thread -> (k row, 16-col segment)
    const int bk  = tid >> 3;
    const int bn0 = (tid & 7) * 16;

    const unsigned short* aGlob = Abf + (size_t)(rowBase + ar) * D_MODEL + apart * 16;
    unsigned short* aLds0 = &Alds[0][ar * LDS_STRIDE + apart * 16];
    unsigned short* aLds1 = &Alds[1][ar * LDS_STRIDE + apart * 16];
    const float* bGlobBase = we + ((size_t)e * D_MODEL + bk) * D_FF + colBase + bn0;

    // ---- prologue: stage tile kt=0 into buffer 0 ----
    {
        async_copy32_to_lds(aGlob, aLds0);
        const float4* src = (const float4*)bGlobBase;
        float4 f0 = src[0], f1 = src[1], f2 = src[2], f3 = src[3];
        float fv[16] = {f0.x, f0.y, f0.z, f0.w, f1.x, f1.y, f1.z, f1.w,
                        f2.x, f2.y, f2.z, f2.w, f3.x, f3.y, f3.z, f3.w};
#pragma unroll
        for (int i = 0; i < 16; ++i) {
            Blds[0][(bn0 + i) * LDS_STRIDE + bk] =
                __builtin_bit_cast(unsigned short, (__bf16)fv[i]);
        }
        wait_asynccnt0();
        __syncthreads();
    }

    int buf = 0;
    for (int kt = 0; kt < D_MODEL; kt += 32) {
        const bool haveNext = (kt + 32 < D_MODEL);
        float4 nf0, nf1, nf2, nf3;
        if (haveNext) {
            // issue next-tile VMEM early: B globals into regs, A async straight to LDS
            const float4* src = (const float4*)(bGlobBase + (size_t)(kt + 32) * D_FF);
            nf0 = src[0]; nf1 = src[1]; nf2 = src[2]; nf3 = src[3];
            if (kt + 64 < D_MODEL) {
                __builtin_prefetch(bGlobBase + (size_t)(kt + 64) * D_FF, 0, 1);
            }
            async_copy32_to_lds(aGlob + kt + 32, (buf ? aLds0 : aLds1));
        }

        // ---- fragment loads from current buffer ----
        // ISA 16-bit A/B layout: lanes 0-15 hold K{0..7,16..23}, lanes 16-31 hold K{8..15,24..31}
        union Frag { uint4 u[2]; v16bf v; };
        Frag af[2], bf_[4];
#pragma unroll
        for (int mi = 0; mi < 2; ++mi) {
            const uint4* ap = (const uint4*)(&Alds[buf][(waveM * 32 + mi * 16 + rsel) * LDS_STRIDE + hoff]);
            af[mi].u[0] = ap[0];   // K 0..7   (or 8..15)
            af[mi].u[1] = ap[2];   // K 16..23 (or 24..31)  (+32B)
        }
#pragma unroll
        for (int ni = 0; ni < 4; ++ni) {
            const uint4* bp = (const uint4*)(&Blds[buf][(waveN * 64 + ni * 16 + rsel) * LDS_STRIDE + hoff]);
            bf_[ni].u[0] = bp[0];
            bf_[ni].u[1] = bp[2];
        }

        // ---- next-tile B conversion into the other buffer (overlaps with WMMA) ----
        if (haveNext) {
            float fv[16] = {nf0.x, nf0.y, nf0.z, nf0.w, nf1.x, nf1.y, nf1.z, nf1.w,
                            nf2.x, nf2.y, nf2.z, nf2.w, nf3.x, nf3.y, nf3.z, nf3.w};
            unsigned short* bl = &Blds[buf ^ 1][0];
#pragma unroll
            for (int i = 0; i < 16; ++i) {
                bl[(bn0 + i) * LDS_STRIDE + bk] =
                    __builtin_bit_cast(unsigned short, (__bf16)fv[i]);
            }
        }

        // ---- 8 WMMAs on current buffer ----
#pragma unroll
        for (int mi = 0; mi < 2; ++mi) {
#pragma unroll
            for (int ni = 0; ni < 4; ++ni) {
                acc[mi][ni] = __builtin_amdgcn_wmma_f32_16x16x32_bf16(
                    false, af[mi].v, false, bf_[ni].v, (short)0, acc[mi][ni], false, false);
            }
        }

        wait_asynccnt0();     // this wave's async A-tile transfers done
        __syncthreads();      // all waves' LDS writes visible
        buf ^= 1;
    }

    // ---- epilogue: +bias, store fp32 per ISA C/D layout ----
#pragma unroll
    for (int mi = 0; mi < 2; ++mi) {
        const int rbase = rowBase + waveM * 32 + mi * 16 + (low ? 0 : 8);
#pragma unroll
        for (int ni = 0; ni < 4; ++ni) {
            const int col = colBase + waveN * 64 + ni * 16 + (lane & 15);
            const float bias = be[e * D_FF + col];
#pragma unroll
            for (int i = 0; i < 8; ++i) {
                out[(size_t)(rbase + i) * D_FF + col] = acc[mi][ni][i] + bias;
            }
        }
    }
}

extern "C" void kernel_launch(void* const* d_in, const int* in_sizes, int n_in,
                              void* d_out, int out_size, void* d_ws, size_t ws_size,
                              hipStream_t stream) {
    const float* x  = (const float*)d_in[0];   // [4,2048,2048]
    const float* wg = (const float*)d_in[1];   // [2048,8]
    const float* we = (const float*)d_in[2];   // [8,2048,8192]
    const float* be = (const float*)d_in[3];   // [8,8192]
    float* out = (float*)d_out;                // [8192,8192]

    char* ws = (char*)d_ws;
    int* idx              = (int*)(ws);                       // 32 KB
    int* slot_src         = (int*)(ws + 32768);               // 32 KB
    unsigned short* Abf   = (unsigned short*)(ws + 65536);    // 32 MB: [E*C, D_MODEL] bf16

    moe_gate    <<<NTOK / 8, 256, 0, stream>>>(x, wg, idx);
    moe_assign  <<<1, 1024, 0, stream>>>(idx, slot_src);
    moe_dispatch<<<NTOK, 256, 0, stream>>>(x, slot_src, Abf);
    moe_gemm    <<<dim3(D_FF / 128, CAP / 128, NEXP), 256, 0, stream>>>(Abf, we, be, out);
}
